// SimpleMemoryBank_850403525338
// MI455X (gfx1250) — matrix operand order repaired and checked by reading, not compiled, not run on hardware
//
#include <hip/hip_runtime.h>

typedef __attribute__((ext_vector_type(16))) __bf16 v16bf;
typedef __attribute__((ext_vector_type(8)))  float  v8f;

constexpr int kB = 8;
constexpr int kT = 4096;
constexpr int kD = 512;
constexpr int kS = 128;       // slots
constexpr int kTopK = 8;
constexpr int kRowsPerBlock = 128;

constexpr int kKStride = 520;  // halves per K row in LDS (512 + 8 pad -> bank spread)
constexpr int kSStride = 132;  // floats per score row in LDS (128 + 4 pad)

constexpr int kKOff   = 0;
constexpr int kScOff  = kS * kKStride * 2;                      // 133120 B
constexpr int kSalOff = kScOff + kRowsPerBlock * kSStride * 4;  // 200704
constexpr int kTkwOff = kSalOff + kS * 4;                       // 201216
constexpr int kTkiOff = kTkwOff + kRowsPerBlock * kTopK * 4;    // 205312
constexpr int kLdsTotal = kTkiOff + kRowsPerBlock * kTopK * 4;  // 209408 B

__global__ __launch_bounds__(256, 1)
void membank_topk_attn_kernel(const float* __restrict__ q,
                              const float* __restrict__ Km,
                              const float* __restrict__ Vm,
                              const float* __restrict__ sal,
                              float* __restrict__ out_read,
                              float* __restrict__ out_w) {
  extern __shared__ char smem[];
  unsigned short* kLds  = (unsigned short*)(smem + kKOff);
  float*          scLds = (float*)(smem + kScOff);
  float*          salLds= (float*)(smem + kSalOff);
  float*          tkw   = (float*)(smem + kTkwOff);
  int*            tki   = (int*)(smem + kTkiOff);

  const int tid      = threadIdx.x;
  const int blockRow = blockIdx.x * kRowsPerBlock;

  // ---- Phase 0: K (f32) -> LDS (bf16 via native packed cvt); salience -> LDS ----
  for (int c = tid; c < (kS * kD) / 4; c += 256) {
    int row = c >> 7;            // 128 float4 chunks per row
    int col = (c & 127) << 2;
    float4 v = *(const float4*)(Km + (size_t)row * kD + col);
    union { __bf16 h[4]; uint2 u; } pk;
    pk.h[0] = (__bf16)v.x; pk.h[1] = (__bf16)v.y;
    pk.h[2] = (__bf16)v.z; pk.h[3] = (__bf16)v.w;
    *(uint2*)(kLds + row * kKStride + col) = pk.u;
  }
  if (tid < kS) salLds[tid] = sal[tid];
  __syncthreads();

  // ---- Phase 1: WMMA scores. Each wave computes 16 rows x 128 slots ----
  const int lane = tid & 31;
  const int wave = tid >> 5;
  const int nIdx = lane & 15;   // A: M index; B: N index; C/D: N index
  const int hi16 = lane >> 4;   // half-wave selector

  v8f acc[8];
  const v8f vzero = {0.f, 0.f, 0.f, 0.f, 0.f, 0.f, 0.f, 0.f};
#pragma unroll
  for (int s = 0; s < 8; ++s) acc[s] = vzero;

  const float* qr = q + (size_t)(blockRow + wave * 16 + nIdx) * kD;

  for (int kc = 0; kc < 16; ++kc) {
    // A fragment: lane group hi16 picks K base +0/+16;
    // halves 0-15 <- K[base..base+15] (contiguous, matches B).
    const int kbase = kc * 32 + hi16 * 16;
    float4 a0 = *(const float4*)(qr + kbase);
    float4 a1 = *(const float4*)(qr + kbase + 4);
    float4 a2 = *(const float4*)(qr + kbase + 8);
    float4 a3 = *(const float4*)(qr + kbase + 12);
    union { __bf16 h[16]; v16bf v; } A;
    A.h[0]  = (__bf16)a0.x; A.h[1]  = (__bf16)a0.y; A.h[2]  = (__bf16)a0.z; A.h[3]  = (__bf16)a0.w;
    A.h[4]  = (__bf16)a1.x; A.h[5]  = (__bf16)a1.y; A.h[6]  = (__bf16)a1.z; A.h[7]  = (__bf16)a1.w;
    A.h[8]  = (__bf16)a2.x; A.h[9]  = (__bf16)a2.y; A.h[10] = (__bf16)a2.z; A.h[11] = (__bf16)a2.w;
    A.h[12] = (__bf16)a3.x; A.h[13] = (__bf16)a3.y; A.h[14] = (__bf16)a3.z; A.h[15] = (__bf16)a3.w;

    // B: lanes 0-15 hold K 0-15, lanes 16-31 hold K 16-31 for this k-chunk.
    const int kb = kc * 32 + hi16 * 16;
    // Two bases so every ds offset (s'*16640 B) fits the 16-bit DS immediate.
    const unsigned short* bpLo = kLds + (size_t)nIdx * kKStride + kb;
    const unsigned short* bpHi = bpLo + 64 * kKStride;  // + slot-tile 4 (4*16*520 halves)

    union BFrag { uint4 q2[2]; v16bf v; };
    BFrag Bf[8];
#pragma unroll
    for (int s = 0; s < 4; ++s) {
      const unsigned short* bp = bpLo + (size_t)s * 16 * kKStride;
      Bf[s].q2[0] = *(const uint4*)(bp);
      Bf[s].q2[1] = *(const uint4*)(bp + 8);
    }
#pragma unroll
    for (int s = 0; s < 4; ++s) {
      const unsigned short* bp = bpHi + (size_t)s * 16 * kKStride;
      Bf[s + 4].q2[0] = *(const uint4*)(bp);
      Bf[s + 4].q2[1] = *(const uint4*)(bp + 8);
    }
    // Back-to-back WMMAs: distinct accumulators, shared A, no RAW hazards.
#pragma unroll
    for (int s = 0; s < 8; ++s) {
      acc[s] = __builtin_amdgcn_wmma_f32_16x16x32_bf16(
          false, A.v, false, Bf[s].v, (short)0, acc[s], false, false);
    }
  }

  // Writeback: C/D layout -> VGPR r holds row (r + 8*hi16), lane%16 holds N.
  const float scale = 0.044194173824159216f;  // 1/sqrt(512)
#pragma unroll
  for (int s = 0; s < 8; ++s) {
    float salv = salLds[s * 16 + nIdx];
#pragma unroll
    for (int r = 0; r < 8; ++r) {
      int rowL = wave * 16 + r + 8 * hi16;
      scLds[rowL * kSStride + s * 16 + nIdx] = acc[s][r] * scale + salv;
    }
  }
  __syncthreads();

  // ---- Phase 2: per-row top-8 (descending, lowest-index tie-break) + softmax ----
  if (tid < kRowsPerBlock) {
    float* sc = scLds + tid * kSStride;
    float w[kTopK]; int id[kTopK];
#pragma unroll 1
    for (int j = 0; j < kTopK; ++j) {
      float best = -3.4e38f; int bi = 0;
      for (int i = 0; i < kS; ++i) {
        float v = sc[i];
        if (v > best) { best = v; bi = i; }
      }
      w[j] = best; id[j] = bi;
      sc[bi] = -3.4e38f;  // destructive mark
    }
    float m = w[0];       // first pick is the max
    float e[kTopK]; float sum = 0.f;
#pragma unroll
    for (int j = 0; j < kTopK; ++j) { e[j] = __expf(w[j] - m); sum += e[j]; }
    float inv = 1.0f / sum;
    int grow = blockRow + tid;
#pragma unroll
    for (int j = 0; j < kTopK; ++j) {
      float wj = e[j] * inv;
      out_w[(size_t)grow * kTopK + j] = wj;
      tkw[tid * kTopK + j] = wj;
      tki[tid * kTopK + j] = id[j];
    }
  }
  __syncthreads();

  // ---- Phase 3: read_vectors = sum_j w_j * V[idx_j]; 2 threads per row ----
  {
    const int rowL = tid >> 1;
    const int half = tid & 1;
    const int grow = blockRow + rowL;
    const float4* vrow[kTopK];
    float wj[kTopK];
#pragma unroll
    for (int j = 0; j < kTopK; ++j) {
      vrow[j] = (const float4*)(Vm + (size_t)tki[rowL * kTopK + j] * kD);
      wj[j]   = tkw[rowL * kTopK + j];
    }
    float4* dst = (float4*)(out_read + (size_t)grow * kD);
    for (int c = 0; c < 64; ++c) {
      int cc = half * 64 + c;
      float4 a = {0.f, 0.f, 0.f, 0.f};
#pragma unroll
      for (int j = 0; j < kTopK; ++j) {
        float4 v = vrow[j][cc];
        a.x += wj[j] * v.x; a.y += wj[j] * v.y;
        a.z += wj[j] * v.z; a.w += wj[j] * v.w;
      }
      dst[cc] = a;
    }
  }
}

extern "C" void kernel_launch(void* const* d_in, const int* in_sizes, int n_in,
                              void* d_out, int out_size, void* d_ws, size_t ws_size,
                              hipStream_t stream) {
  const float* q   = (const float*)d_in[0];
  const float* Km  = (const float*)d_in[1];
  const float* Vm  = (const float*)d_in[2];
  const float* sal = (const float*)d_in[3];
  // d_in[4] = topk scalar (fixed at 8 here)

  float* out_read = (float*)d_out;
  float* out_w    = out_read + (size_t)kB * kT * kD;

  dim3 grid((kB * kT) / kRowsPerBlock);  // 256 blocks
  dim3 block(256);                        // 8 waves (wave32)
  membank_topk_attn_kernel<<<grid, block, kLdsTotal, stream>>>(
      q, Km, Vm, sal, out_read, out_w);
}
